// HashEmbedder_44968307589316
// MI455X (gfx1250) — compile-verified, matrix operand order checked by reading, not run on hardware
//
#include <hip/hip_runtime.h>
#include <hip/hip_bf16.h>

typedef __attribute__((ext_vector_type(16))) _Float16 v16h;
typedef __attribute__((ext_vector_type(8)))  _Float16 v8h;
typedef __attribute__((ext_vector_type(8)))  float    v8f;

#define BATCH      131072
#define NLEV       16
#define HID        512
#define HASH_MASK  ((1u << 19) - 1u)
#define PI_F       3.14159265358979323846f

#define ROWS_PER_BLOCK 128
#define THREADS        512   // 16 waves
#define LDS_STRIDE     520   // 512 + 8 halves pad -> conflict-free column stores

// packed-weight layout in d_ws (units: _Float16 halves)
#define OFF_W1 0
#define SZ_W1  (32 * 512)            // NT=32, KT=1 (K padded 5->32)
#define SZ_W   (32 * 16 * 512)       // NT=32, KT=16  (512x512)
#define OFF_W2 (OFF_W1 + SZ_W1)
#define OFF_W3 (OFF_W2 + SZ_W)
#define OFF_W4 (OFF_W3 + SZ_W)
#define OFF_W5 (OFF_W4 + SZ_W)
#define OFF_W6 (OFF_W5 + SZ_W)
#define SZ_W6  (16 * 512)            // NT=1, KT=16   (512x16)

__device__ __constant__ float RES_TAB[16] = {
    16.f, 20.f, 25.f, 32.f, 40.f, 50.f, 64.f, 80.f,
    101.f, 128.f, 161.f, 203.f, 256.f, 322.f, 406.f, 512.f};

__device__ __constant__ float BOX_MIN_C[4] = {0.f, -PI_F, 0.5f * PI_F, -0.85f * PI_F};
__device__ __constant__ float BOX_MAX_C[4] = {PI_F, PI_F, 0.85f * PI_F, -0.5f * PI_F};

// ---------------------------------------------------------------------------
// Pack a (Ksrc x Nsrc) row-major f32 weight matrix into f16 WMMA B-fragments.
// Tile (nt, kt) covers K = kt*32..+31, N = nt*16..+15; 512 halves per tile,
// per-lane layout from CDNA5 ISA 7.12.2 (16-bit 16x32 matrix striping).
// ---------------------------------------------------------------------------
__global__ __launch_bounds__(256) void pack_weights_kernel(
    const float* __restrict__ W, _Float16* __restrict__ dst,
    int Ksrc, int Nsrc, int KT, int NT) {
  int gid = blockIdx.x * 256 + threadIdx.x;
  int total = NT * KT * 32 * 16;
  if (gid >= total) return;
  int j    = gid & 15;          // element within lane's 16 halves
  int lane = (gid >> 4) & 31;
  int tile = gid >> 9;          // tile = nt*KT + kt
  int kt = tile % KT;
  int nt = tile / KT;
  int v = j >> 1, p = j & 1;
  int k = kt * 32 + ((v >= 4) ? 16 : 0) + (v & 3) * 2 + p + ((lane >= 16) ? 8 : 0);
  int n = nt * 16 + (lane & 15);
  float val = (k < Ksrc && n < Nsrc) ? W[(size_t)k * Nsrc + n] : 0.f;
  dst[(size_t)tile * 512 + (size_t)lane * 16 + j] = (_Float16)val;
}

// ---------------------------------------------------------------------------
// A-fragment load from LDS activation buffer (row-major f16, LDS_STRIDE/row).
// Lane layout (16-bit A 16x32): first 8 halves = K kbase..kbase+7,
// next 8 = kbase+16..kbase+23; kbase includes the +8 for lanes 16..31.
// ---------------------------------------------------------------------------
__device__ __forceinline__ v16h load_a_frag(const _Float16* hbuf, int m, int kbase) {
  const v8h lo = *(const v8h*)(hbuf + (size_t)m * LDS_STRIDE + kbase);
  const v8h hi = *(const v8h*)(hbuf + (size_t)m * LDS_STRIDE + kbase + 16);
  v16h a;
#pragma unroll
  for (int j = 0; j < 8; ++j) { a[j] = lo[j]; a[8 + j] = hi[j]; }
  return a;
}

// C/D tile (16x16 f32, 8 VGPRs): lane l holds column n = l&15,
// rows m = r + 8*(l>=16). Fuse bias + ReLU, store f16 into LDS.
__device__ __forceinline__ void store_tile_relu(_Float16* hbuf, v8f c, int mt, int nt,
                                                int lane, const float* __restrict__ bias) {
  int n  = nt * 16 + (lane & 15);
  float bv = bias[n];
  int mbase = mt * 16 + ((lane >> 4) << 3);
#pragma unroll
  for (int r = 0; r < 8; ++r) {
    float vv = c[r] + bv;
    vv = vv > 0.f ? vv : 0.f;
    hbuf[(size_t)(mbase + r) * LDS_STRIDE + n] = (_Float16)vv;
  }
}

// ---------------------------------------------------------------------------
// Fused per-level kernel: 128 points/block, 16 waves.
// wave w: mt = w>>1 (16-row tile 0..7), ntiles (w&1)*16 .. +15 of the 512 cols.
// ---------------------------------------------------------------------------
__global__ __launch_bounds__(THREADS) void hash_mlp_kernel(
    const float* __restrict__ x, const float* __restrict__ tables,
    const _Float16* __restrict__ wp,
    const float* __restrict__ b1, const float* __restrict__ b2,
    const float* __restrict__ b3, const float* __restrict__ b4,
    const float* __restrict__ b5, const float* __restrict__ b6,
    float* __restrict__ out) {
  __shared__ _Float16 h_lds[ROWS_PER_BLOCK * LDS_STRIDE];  // 130 KB

  const int tid  = threadIdx.x;
  const int lane = tid & 31;
  const int wave = tid >> 5;
  const int mt   = wave >> 1;          // 0..7
  const int ntb  = (wave & 1) * 16;    // 0 or 16
  const int lev  = blockIdx.y;
  const int row0 = blockIdx.x * ROWS_PER_BLOCK;
  const float res = RES_TAB[lev];

  const int mrow = mt * 16 + (lane & 15);     // row within block for A frags
  const int khi  = (lane >> 4) * 8;           // +8 K offset for upper lanes

  // ---------------- layer 1: feat=[res,x] (K padded to 32), one WMMA step ----
  v16h a1;
#pragma unroll
  for (int j = 0; j < 16; ++j) a1[j] = (_Float16)0.f;
  if (lane < 16) {
    const float4 xv = ((const float4*)x)[row0 + mrow];
    a1[0] = (_Float16)res;
    a1[1] = (_Float16)xv.x;
    a1[2] = (_Float16)xv.y;
    a1[3] = (_Float16)xv.z;
    a1[4] = (_Float16)xv.w;
  }
  {
    v8f C[16];
#pragma unroll
    for (int i = 0; i < 16; ++i)
#pragma unroll
      for (int r = 0; r < 8; ++r) C[i][r] = 0.f;
    const _Float16* w1 = wp + OFF_W1 + (size_t)lane * 16;
#pragma unroll
    for (int i = 0; i < 16; ++i) {
      v16h bf = *(const v16h*)(w1 + (size_t)(ntb + i) * 512);
      C[i] = __builtin_amdgcn_wmma_f32_16x16x32_f16(false, a1, false, bf,
                                                    (short)0, C[i], false, false);
    }
#pragma unroll
    for (int i = 0; i < 16; ++i) store_tile_relu(h_lds, C[i], mt, ntb + i, lane, b1);
  }
  __syncthreads();

  // ---------------- hidden layers 2..5 (512x512) -----------------------------
  const _Float16* wbases[4] = {wp + OFF_W2, wp + OFF_W3, wp + OFF_W4, wp + OFF_W5};
  const float* biases[4] = {b2, b3, b4, b5};
#pragma unroll 1
  for (int layer = 0; layer < 4; ++layer) {
    const _Float16* wbase = wbases[layer] + (size_t)lane * 16;
    v8f C[16];
#pragma unroll
    for (int i = 0; i < 16; ++i)
#pragma unroll
      for (int r = 0; r < 8; ++r) C[i][r] = 0.f;
#pragma unroll 1
    for (int kt = 0; kt < 16; ++kt) {
      const v16h a = load_a_frag(h_lds, mrow, kt * 32 + khi);
      const _Float16* wk = wbase + (size_t)kt * 512;
#pragma unroll
      for (int i = 0; i < 16; ++i) {
        v16h bf = *(const v16h*)(wk + (size_t)(ntb + i) * (16 * 512));
        C[i] = __builtin_amdgcn_wmma_f32_16x16x32_f16(false, a, false, bf,
                                                      (short)0, C[i], false, false);
      }
    }
    __syncthreads();  // everyone done reading this layer's activations
#pragma unroll
    for (int i = 0; i < 16; ++i)
      store_tile_relu(h_lds, C[i], mt, ntb + i, lane, biases[layer]);
    __syncthreads();
  }

  // ---------------- layer 6: 512 -> 16 logits (waves 0..7 only) --------------
  v8f c6;
#pragma unroll
  for (int r = 0; r < 8; ++r) c6[r] = 0.f;
  if (wave < 8) {
    const _Float16* w6 = wp + OFF_W6 + (size_t)lane * 16;
    const int m6 = wave * 16 + (lane & 15);
#pragma unroll 1
    for (int kt = 0; kt < 16; ++kt) {
      const v16h a = load_a_frag(h_lds, m6, kt * 32 + khi);
      v16h bf = *(const v16h*)(w6 + (size_t)kt * 512);
      c6 = __builtin_amdgcn_wmma_f32_16x16x32_f16(false, a, false, bf,
                                                  (short)0, c6, false, false);
    }
  }
  __syncthreads();  // all reads of h5 done -> safe to reuse LDS for logits
  if (wave < 8) {
    float* lg = (float*)h_lds;  // 128 x 16 f32 logits
    const int n = lane & 15;
    const int mbase = wave * 16 + ((lane >> 4) << 3);
    const float bv = b6[n];
#pragma unroll
    for (int r = 0; r < 8; ++r) lg[(mbase + r) * 16 + n] = c6[r] + bv;
  }
  __syncthreads();

  // ---------------- softmax + hash-grid gather (one thread per row) ----------
  if (tid < ROWS_PER_BLOCK) {
    const float* lg = (const float*)h_lds;
    const int b = row0 + tid;
    float w[16];
    float mx = -3.4e38f;
#pragma unroll
    for (int n = 0; n < 16; ++n) mx = fmaxf(mx, lg[tid * 16 + n]);
    float s = 0.f;
#pragma unroll
    for (int n = 0; n < 16; ++n) { w[n] = __expf(lg[tid * 16 + n] - mx); s += w[n]; }
    const float inv = 1.f / s;

    int bl[4];
#pragma unroll
    for (int d = 0; d < 4; ++d) {
      float xv = x[(size_t)b * 4 + d];
      float xc = fminf(fmaxf(xv, BOX_MIN_C[d]), BOX_MAX_C[d]);
      float grid = (BOX_MAX_C[d] - BOX_MIN_C[d]) / res;
      bl[d] = (int)floorf((xc - BOX_MIN_C[d]) / grid);
    }
    const float* tb = tables + (size_t)lev * ((size_t)1 << 19) * 2;
    float o0 = 0.f, o1 = 0.f;
#pragma unroll
    for (int v = 0; v < 16; ++v) {
      unsigned h = (unsigned)(bl[0] + ((v >> 3) & 1));
      h ^= (unsigned)(bl[1] + ((v >> 2) & 1)) * 2654435761u;
      h ^= (unsigned)(bl[2] + ((v >> 1) & 1)) * 805459861u;
      h ^= (unsigned)(bl[3] + (v & 1)) * 3674653429u;
      unsigned idx = h & HASH_MASK;
      float wv = w[v] * inv;
      o0 += wv * tb[(size_t)idx * 2 + 0];
      o1 += wv * tb[(size_t)idx * 2 + 1];
    }
    out[(size_t)b * 32 + lev * 2 + 0] = o0;
    out[(size_t)b * 32 + lev * 2 + 1] = o1;
  }
}

// ---------------------------------------------------------------------------
// keep_mask = all(x == clip(x, box)) written as 0.0/1.0 after the (B,32) block
// ---------------------------------------------------------------------------
__global__ __launch_bounds__(256) void mask_kernel(const float* __restrict__ x,
                                                   float* __restrict__ outm) {
  int t = blockIdx.x * 256 + threadIdx.x;
  if (t >= BATCH) return;
  bool keep = true;
#pragma unroll
  for (int d = 0; d < 4; ++d) {
    float v = x[(size_t)t * 4 + d];
    keep = keep && (v >= BOX_MIN_C[d]) && (v <= BOX_MAX_C[d]);
  }
  outm[t] = keep ? 1.f : 0.f;
}

extern "C" void kernel_launch(void* const* d_in, const int* in_sizes, int n_in,
                              void* d_out, int out_size, void* d_ws, size_t ws_size,
                              hipStream_t stream) {
  const float* x      = (const float*)d_in[0];
  const float* tables = (const float*)d_in[1];
  const float* W1 = (const float*)d_in[2];  const float* b1 = (const float*)d_in[3];
  const float* W2 = (const float*)d_in[4];  const float* b2 = (const float*)d_in[5];
  const float* W3 = (const float*)d_in[6];  const float* b3 = (const float*)d_in[7];
  const float* W4 = (const float*)d_in[8];  const float* b4 = (const float*)d_in[9];
  const float* W5 = (const float*)d_in[10]; const float* b5 = (const float*)d_in[11];
  const float* W6 = (const float*)d_in[12]; const float* b6 = (const float*)d_in[13];
  float* out = (float*)d_out;
  _Float16* wp = (_Float16*)d_ws;

  // Pack weights into WMMA B-fragment layout (idempotent, runs every call).
  {
    int t1 = 32 * 1 * 512;   // W1 tiles * 512
    pack_weights_kernel<<<(t1 + 255) / 256, 256, 0, stream>>>(W1, wp + OFF_W1, 5, 512, 1, 32);
    int th = 32 * 16 * 512;
    pack_weights_kernel<<<(th + 255) / 256, 256, 0, stream>>>(W2, wp + OFF_W2, 512, 512, 16, 32);
    pack_weights_kernel<<<(th + 255) / 256, 256, 0, stream>>>(W3, wp + OFF_W3, 512, 512, 16, 32);
    pack_weights_kernel<<<(th + 255) / 256, 256, 0, stream>>>(W4, wp + OFF_W4, 512, 512, 16, 32);
    pack_weights_kernel<<<(th + 255) / 256, 256, 0, stream>>>(W5, wp + OFF_W5, 512, 512, 16, 32);
    int t6 = 1 * 16 * 512;
    pack_weights_kernel<<<(t6 + 255) / 256, 256, 0, stream>>>(W6, wp + OFF_W6, 512, 16, 16, 1);
  }

  dim3 grid(BATCH / ROWS_PER_BLOCK, NLEV);
  hash_mlp_kernel<<<grid, THREADS, 0, stream>>>(x, tables, wp, b1, b2, b3, b4, b5, b6, out);

  mask_kernel<<<BATCH / 256, 256, 0, stream>>>(x, out + (size_t)BATCH * 32);
}
